// SoftDecisionTree_73555609911421
// MI455X (gfx1250) — matrix operand order repaired and checked by reading, not compiled, not run on hardware
//
#include <hip/hip_runtime.h>

typedef __attribute__((ext_vector_type(16))) __bf16 v16bf;
typedef __attribute__((ext_vector_type(8)))  __bf16 v8bf;
typedef __attribute__((ext_vector_type(8)))  float  v8f;

#define BROWS   8192
#define DDIM    4096
#define NNODES  1023
#define NLEAVES 1024

#define MT      64      // rows per block
#define NCH     128     // node columns per chunk
#define KCH     64      // K per LDS stage
#define THREADS 512
#define NWAVES  16

#define A_STRIDE 72     // bf16 elems per LDS row (64 + 8 pad) = 144 B
#define B_STRIDE 72

// ---------------------------------------------------------------------------
// Kernel 1: leaf_val[i] = softmax(leaf_dist[i,:]) . class_reward
// ---------------------------------------------------------------------------
__global__ void leafval_kernel(const float* __restrict__ leaf_dist,
                               const float* __restrict__ class_reward,
                               float* __restrict__ leaf_val) {
    int i = blockIdx.x * blockDim.x + threadIdx.x;
    if (i < NLEAVES) {
        float a = leaf_dist[i * 2 + 0];
        float b = leaf_dist[i * 2 + 1];
        float m = fmaxf(a, b);
        float e0 = __expf(a - m);
        float e1 = __expf(b - m);
        float inv = 1.0f / (e0 + e1);
        leaf_val[i] = (e0 * class_reward[0] + e1 * class_reward[1]) * inv;
    }
}

// ---------------------------------------------------------------------------
// helpers
// ---------------------------------------------------------------------------
__device__ __forceinline__ v8bf cvt8(float4 f0, float4 f1, float s) {
    v8bf t;
    t[0] = (__bf16)(f0.x * s); t[1] = (__bf16)(f0.y * s);
    t[2] = (__bf16)(f0.z * s); t[3] = (__bf16)(f0.w * s);
    t[4] = (__bf16)(f1.x * s); t[5] = (__bf16)(f1.y * s);
    t[6] = (__bf16)(f1.z * s); t[7] = (__bf16)(f1.w * s);
    return t;
}

__device__ __forceinline__ v16bf ldfrag(const __bf16* p, int secondOff) {
    v8bf lo = *(const v8bf*)(p);
    v8bf hi = *(const v8bf*)(p + secondOff);
    return __builtin_shufflevector(lo, hi,
        0, 1, 2, 3, 4, 5, 6, 7, 8, 9, 10, 11, 12, 13, 14, 15);
}

// ---------------------------------------------------------------------------
// Kernel 2: fused  sigmoid(beta*(x@W.T+b)) -> tree path product -> @leaf_val
// ---------------------------------------------------------------------------
__launch_bounds__(THREADS, 1)
__global__ void softtree_kernel(const float* __restrict__ x,
                                const float* __restrict__ W,
                                const float* __restrict__ bias,
                                const float* __restrict__ beta,
                                const float* __restrict__ leaf_val,
                                float* __restrict__ out) {
    extern __shared__ char smem[];
    float*  probs = (float*)smem;                                        // MT x 1024 f32
    __bf16* As    = (__bf16*)(smem + MT * 1024 * 4);                     // MT x A_STRIDE
    __bf16* Bs    = (__bf16*)(smem + MT * 1024 * 4 + MT * A_STRIDE * 2); // NCH x B_STRIDE

    const int tid     = threadIdx.x;
    const int lane    = tid & 31;
    const int wv      = tid >> 5;
    const int rowBase = blockIdx.x * MT;

    const int mt   = wv & 3;          // which 16-row M tile (0..3)
    const int ntb  = (wv >> 2) * 2;   // base 16-col N tile (2 tiles per wave)
    const int half = lane >> 4;
    const int l15  = lane & 15;

    // staging assignments (loop invariant)
    const int ar = tid >> 3;                 // A: row 0..63
    const int ac = (tid & 7) * 8;            // A: col 0..56
    const int bn = tid >> 2;                 // B: node-row 0..127
    const int bc = (tid & 3) * 16;           // B: col 0,16,32,48
    const float* srcA = x + (size_t)(rowBase + ar) * DDIM + ac;
    __bf16* dstA = As + ar * A_STRIDE + ac;
    __bf16* dstB = Bs + bn * B_STRIDE + bc;

    // fragment base pointers (loop invariant)
    const __bf16* apB  = As + (mt * 16 + l15) * A_STRIDE + half * 16;
    const __bf16* bp0B = Bs + ((ntb + 0) * 16 + l15) * B_STRIDE + half * 16;
    const __bf16* bp1B = Bs + ((ntb + 1) * 16 + l15) * B_STRIDE + half * 16;

    for (int nc = 0; nc < 1024 / NCH; ++nc) {
        const int nbase = nc * NCH;
        const int node  = nbase + bn;
        const float bMask = (node < NNODES) ? 1.0f : 0.0f;   // uniform flow, no exec branch
        const float* srcB = W + (size_t)min(node, NNODES - 1) * DDIM + bc;

        v8f acc0 = {0.f, 0.f, 0.f, 0.f, 0.f, 0.f, 0.f, 0.f};
        v8f acc1 = {0.f, 0.f, 0.f, 0.f, 0.f, 0.f, 0.f, 0.f};

        // ---- register prefetch of k0 = 0 slice
        float4 pa0 = *(const float4*)(srcA + 0);
        float4 pa1 = *(const float4*)(srcA + 4);
        float4 pb0 = *(const float4*)(srcB + 0);
        float4 pb1 = *(const float4*)(srcB + 4);
        float4 pb2 = *(const float4*)(srcB + 8);
        float4 pb3 = *(const float4*)(srcB + 12);

        for (int k0 = 0; k0 < DDIM; k0 += KCH) {
            // ---- commit prefetched slice to LDS (bf16)
            *(v8bf*)(dstA)     = cvt8(pa0, pa1, 1.0f);
            *(v8bf*)(dstB)     = cvt8(pb0, pb1, bMask);
            *(v8bf*)(dstB + 8) = cvt8(pb2, pb3, bMask);
            __syncthreads();

            // ---- issue global prefetch for next slice; wait lands after WMMAs
            if (k0 + KCH < DDIM) {
                const float* nA = srcA + k0 + KCH;
                const float* nB = srcB + k0 + KCH;
                pa0 = *(const float4*)(nA + 0);
                pa1 = *(const float4*)(nA + 4);
                pb0 = *(const float4*)(nB + 0);
                pb1 = *(const float4*)(nB + 4);
                pb2 = *(const float4*)(nB + 8);
                pb3 = *(const float4*)(nB + 12);
            }

            // ---- load all six fragments, then issue four WMMAs
            v16bf a0  = ldfrag(apB, 8);           // kk = 0
            v16bf b00 = ldfrag(bp0B, 8);
            v16bf b01 = ldfrag(bp1B, 8);
            v16bf a1  = ldfrag(apB + 32, 8);      // kk = 32
            v16bf b10 = ldfrag(bp0B + 32, 8);
            v16bf b11 = ldfrag(bp1B + 32, 8);

            acc0 = __builtin_amdgcn_wmma_f32_16x16x32_bf16(
                false, a0, false, b00, (short)0, acc0, false, false);
            acc1 = __builtin_amdgcn_wmma_f32_16x16x32_bf16(
                false, a0, false, b01, (short)0, acc1, false, false);
            acc0 = __builtin_amdgcn_wmma_f32_16x16x32_bf16(
                false, a1, false, b10, (short)0, acc0, false, false);
            acc1 = __builtin_amdgcn_wmma_f32_16x16x32_bf16(
                false, a1, false, b11, (short)0, acc1, false, false);

            __syncthreads();
        }

        // ---- epilogue: p = sigmoid(beta*(acc + b)) -> probs LDS
        #pragma unroll
        for (int ti = 0; ti < 2; ++ti) {
            v8f accv = ti ? acc1 : acc0;
            int ncol  = (ntb + ti) * 16 + l15;
            int onode = nbase + ncol;
            float bb = 0.f, bt = 0.f;
            if (onode < NNODES) { bb = bias[onode]; bt = beta[onode]; }
            #pragma unroll
            for (int r = 0; r < 8; ++r) {
                int m = mt * 16 + half * 8 + r;   // f32 C/D layout: VGPR r -> M=r / M=8+r
                float z = bt * (accv[r] + bb);
                float p = 1.0f / (1.0f + __expf(-z));
                if (onode >= NNODES) p = 0.0f;
                probs[m * 1024 + onode] = p;
            }
        }
    }
    __syncthreads();

    // ---- tree reduction: v[j] = p*v[2j] + (1-p)*v[2j+1], bottom-up
    // lane l holds leaf values 32l..32l+31
    float vleaf[32];
    #pragma unroll
    for (int i = 0; i < 8; ++i) {
        float4 f = *(const float4*)(leaf_val + lane * 32 + i * 4);
        vleaf[4 * i + 0] = f.x; vleaf[4 * i + 1] = f.y;
        vleaf[4 * i + 2] = f.z; vleaf[4 * i + 3] = f.w;
    }

    for (int rr = 0; rr < MT / NWAVES; ++rr) {
        int row = wv * (MT / NWAVES) + rr;
        const float* pr = probs + row * 1024;
        float v[32];
        #pragma unroll
        for (int i = 0; i < 32; ++i) v[i] = vleaf[i];

        // level 9 (512 nodes, base 511): intra-lane
        #pragma unroll
        for (int j = 0; j < 16; ++j) {
            float p = pr[511 + 16 * lane + j];
            v[j] = p * v[2 * j] + (1.0f - p) * v[2 * j + 1];
        }
        #pragma unroll
        for (int j = 0; j < 8; ++j) {
            float p = pr[255 + 8 * lane + j];
            v[j] = p * v[2 * j] + (1.0f - p) * v[2 * j + 1];
        }
        #pragma unroll
        for (int j = 0; j < 4; ++j) {
            float p = pr[127 + 4 * lane + j];
            v[j] = p * v[2 * j] + (1.0f - p) * v[2 * j + 1];
        }
        #pragma unroll
        for (int j = 0; j < 2; ++j) {
            float p = pr[63 + 2 * lane + j];
            v[j] = p * v[2 * j] + (1.0f - p) * v[2 * j + 1];
        }
        {
            float p = pr[31 + lane];
            v[0] = p * v[0] + (1.0f - p) * v[1];
        }
        // levels 4..0: cross-lane via shuffles
        float vv = v[0];
        #pragma unroll
        for (int d = 4; d >= 0; --d) {
            float va = __shfl(vv, 2 * lane, 32);
            float vb = __shfl(vv, 2 * lane + 1, 32);
            float p  = (lane < (1 << d)) ? pr[(1 << d) - 1 + lane] : 0.0f;
            vv = p * va + (1.0f - p) * vb;
        }
        if (lane == 0) out[rowBase + row] = vv;
    }
}

// ---------------------------------------------------------------------------
extern "C" void kernel_launch(void* const* d_in, const int* in_sizes, int n_in,
                              void* d_out, int out_size, void* d_ws, size_t ws_size,
                              hipStream_t stream) {
    const float* x            = (const float*)d_in[0];
    const float* W            = (const float*)d_in[1];
    const float* b            = (const float*)d_in[2];
    const float* beta         = (const float*)d_in[3];
    const float* leaf_dist    = (const float*)d_in[4];
    const float* class_reward = (const float*)d_in[5];
    float* out      = (float*)d_out;
    float* leaf_val = (float*)d_ws;   // 1024 floats of scratch

    leafval_kernel<<<4, 256, 0, stream>>>(leaf_dist, class_reward, leaf_val);

    size_t smem = (size_t)MT * 1024 * 4        // probs
                + (size_t)MT * A_STRIDE * 2    // As
                + (size_t)NCH * B_STRIDE * 2;  // Bs   = 289,792 B (< 320 KB WGP LDS)
    softtree_kernel<<<BROWS / MT, THREADS, smem, stream>>>(x, W, b, beta, leaf_val, out);
}
